// ParameterizedExperts_58136677318850
// MI455X (gfx1250) — matrix operand order repaired and checked by reading, not compiled, not run on hardware
//
#include <hip/hip_runtime.h>
#include <hip/hip_bf16.h>
#include <stdint.h>

// ---------------- problem constants (match reference) ----------------
#define NUM_E   64
#define D_IN    512
#define D_OUT   512
#define TOT_T   131072

// ---------------- tiling ----------------
#define BM 128
#define BN 128
#define BK 32                       // one bf16 WMMA K-depth per stage
#define KP (BK + 8)                 // padded LDS row: 80B -> bank-conflict free, 16B aligned
#define KSTEPS (D_IN / BK)          // 16
#define MAX_MTILES (TOT_T / BM + NUM_E)   // 1088 worst case for ragged tiles
#define NTILES (D_OUT / BN)         // 4

typedef __attribute__((ext_vector_type(16))) __bf16 v16bf;
typedef __attribute__((ext_vector_type(8)))  __bf16 v8bf;
typedef __attribute__((ext_vector_type(4)))  __bf16 v4bf;
typedef __attribute__((ext_vector_type(8)))  float  v8f;

// ---- CDNA5 async global->LDS (bypasses VGPRs; tracked with ASYNCcnt) ----
__device__ __forceinline__ void async_g2l_b128(uint32_t lds_addr, const void* gaddr) {
    asm volatile("global_load_async_to_lds_b128 %0, %1, off"
                 :: "v"(lds_addr), "v"(gaddr) : "memory");
}
__device__ __forceinline__ void wait_asynccnt0() {
    asm volatile("s_wait_asynccnt 0x0" ::: "memory");
}

// =====================================================================
// Kernel 1: ragged tile table from counts. Entry: {e, row0, rows, 0}.
// =====================================================================
__global__ void moe_build_tiles(const int* __restrict__ counts,
                                int4* __restrict__ tiles) {
    if (threadIdx.x != 0 || blockIdx.x != 0) return;
    int idx = 0, off = 0;
    for (int e = 0; e < NUM_E; ++e) {
        int c = counts[e];
        for (int r = 0; r < c; r += BM) {
            int rc = (c - r) < BM ? (c - r) : BM;
            tiles[idx++] = make_int4(e, off + r, rc, 0);
        }
        off += c;
    }
    for (; idx < MAX_MTILES; ++idx) tiles[idx] = make_int4(0, 0, 0, 0);
}

// =====================================================================
// Kernel 2: weight preconvert fp32 (D_OUT,E,D_IN) -> bf16 (E,D_OUT,D_IN).
// 16.7M elems, 8 per thread. Stays L2-resident (192MB) for the GEMM.
// =====================================================================
__global__ __launch_bounds__(256)
void moe_convert_w(const float* __restrict__ w, __bf16* __restrict__ wbf) {
    const size_t o = ((size_t)blockIdx.x * 256 + threadIdx.x) * 8;  // (e,n,k) linear
    const int k = (int)(o & (D_IN - 1));
    const int n = (int)((o >> 9) & (D_OUT - 1));
    const int e = (int)(o >> 18);
    const float* src = w + ((size_t)n * NUM_E + e) * D_IN + k;
    float4 a = *(const float4*)(src);
    float4 b = *(const float4*)(src + 4);
    v8bf r = { (__bf16)a.x, (__bf16)a.y, (__bf16)a.z, (__bf16)a.w,
               (__bf16)b.x, (__bf16)b.y, (__bf16)b.z, (__bf16)b.w };
    *(v8bf*)(wbf + o) = r;
}

// =====================================================================
// Kernel 3: grouped GEMM, bf16 WMMA, f32 accumulate.
// 256 thr = 8 wave32; tile 128x128; double-buffered LDS; fetch-ahead
// pipeline (global->regs 2 stages ahead, regs->LDS 1 stage ahead).
// PRE=true: B arrives via global_load_async_to_lds_b128 from bf16 weights.
// =====================================================================
template<bool PRE>
__global__ __launch_bounds__(256)
void moe_gemm(const float* __restrict__ x,
              const float* __restrict__ w,       // fp32 (D_OUT,E,D_IN), used if !PRE
              const __bf16* __restrict__ wbf,    // bf16 (E,D_OUT,D_IN), used if PRE
              const float* __restrict__ bias,    // (D_OUT, E)
              const int4*  __restrict__ tiles,
              float* __restrict__ out) {
    __shared__ __align__(16) __bf16 Asm[2][BM][KP];  // 20 KB
    __shared__ __align__(16) __bf16 Bsm[2][BN][KP];  // 20 KB (n-major)

    const int4 t    = tiles[blockIdx.x];
    const int  e    = t.x;
    const int  row0 = t.y;
    const int  rows = t.z;
    if (rows == 0) return;
    const int  n0   = blockIdx.y * BN;

    const int tid  = threadIdx.x;
    const int lane = tid & 31;
    const int wave = tid >> 5;
    const int wm   = wave & 3;      // 4 waves along M (32 rows each)
    const int wn   = wave >> 2;     // 2 waves along N (64 cols each)
    const int half = lane >> 4;     // K-half select per 16-bit WMMA layout
    const int l15  = lane & 15;

    // loader mapping: 128 rows x 32 k per stage; thread -> (row, 16-k half)
    const int ldRow  = tid >> 1;                       // 0..127
    const int ldKoff = (tid & 1) * 16;                 // 0 | 16
    const int aSrcRow = (ldRow < rows) ? ldRow : (rows - 1);  // branchless clamp;
    // rows beyond `rows` feed only outputs we never store -> no zero-fill needed.
    const float*  xRow   = x + (size_t)(row0 + aSrcRow) * D_IN + ldKoff;
    const size_t  wStrideN = (size_t)NUM_E * D_IN;
    const float*  wRow   = w   + ((size_t)(n0 + ldRow)) * wStrideN + (size_t)e * D_IN + ldKoff;
    const __bf16* wbfRow = wbf + ((size_t)e * D_OUT + n0 + ldRow) * D_IN;

    v8f acc[2][4];
    #pragma unroll
    for (int i = 0; i < 2; ++i)
        #pragma unroll
        for (int j = 0; j < 4; ++j)
            acc[i][j] = (v8f){0.f, 0.f, 0.f, 0.f, 0.f, 0.f, 0.f, 0.f};

    float4 aR[4];            // A fetch registers (2 stages ahead)
    float4 bR[4];            // B fetch registers (only when !PRE)

    auto fetch = [&](int kt) {
        const int k0 = kt * BK;
        #pragma unroll
        for (int q = 0; q < 4; ++q) {
            aR[q] = *(const float4*)(xRow + k0 + q * 4);
            if (!PRE) bR[q] = *(const float4*)(wRow + k0 + q * 4);
        }
    };
    auto commit = [&](int buf) {
        #pragma unroll
        for (int q = 0; q < 4; ++q) {
            v4bf ab = { (__bf16)aR[q].x, (__bf16)aR[q].y, (__bf16)aR[q].z, (__bf16)aR[q].w };
            *(v4bf*)&Asm[buf][ldRow][ldKoff + q * 4] = ab;
            if (!PRE) {
                v4bf bb = { (__bf16)bR[q].x, (__bf16)bR[q].y, (__bf16)bR[q].z, (__bf16)bR[q].w };
                *(v4bf*)&Bsm[buf][ldRow][ldKoff + q * 4] = bb;
            }
        }
    };
    auto asyncB = [&](int kt, int buf) {
        if (PRE) {
            const int k0 = kt * BK;
            #pragma unroll
            for (int j = 0; j < 2; ++j) {
                const int ko = ldKoff + j * 8;                       // 8 bf16 = 16B chunk
                uint32_t lds = (uint32_t)(uintptr_t)&Bsm[buf][ldRow][ko];
                async_g2l_b128(lds, wbfRow + k0 + ko);
            }
        }
    };

    // ---- prologue ----
    fetch(0);
    commit(0);
    asyncB(0, 0);
    fetch(1);
    if (PRE) wait_asynccnt0();
    __syncthreads();

    int buf = 0;
    for (int kt = 0; kt < KSTEPS; ++kt) {
        if (kt + 1 < KSTEPS) { commit(buf ^ 1); asyncB(kt + 1, buf ^ 1); }
        if (kt + 2 < KSTEPS) fetch(kt + 2);

        // ---- fragments: A 16x32 (lane<16: K{0..7,16..23}; lane>=16: K{8..15,24..31}),
        //      B 32x16 (lanes 0-15: K{0..7,16..23}, lanes 16-31: K{8..15,24..31} of col N=l15) ----
        v16bf afr[2], bfr[4];
        #pragma unroll
        for (int fm = 0; fm < 2; ++fm) {
            const __bf16* ap = &Asm[buf][wm * 32 + fm * 16 + l15][half * 8];
            ((uint4*)&afr[fm])[0] = *(const uint4*)ap;
            ((uint4*)&afr[fm])[1] = *(const uint4*)(ap + 16);
        }
        #pragma unroll
        for (int fn = 0; fn < 4; ++fn) {
            const __bf16* bp = &Bsm[buf][wn * 64 + fn * 16 + l15][half * 8];
            ((uint4*)&bfr[fn])[0] = *(const uint4*)bp;
            ((uint4*)&bfr[fn])[1] = *(const uint4*)(bp + 16);
        }
        #pragma unroll
        for (int fn = 0; fn < 4; ++fn)
            #pragma unroll
            for (int fm = 0; fm < 2; ++fm)
                acc[fm][fn] = __builtin_amdgcn_wmma_f32_16x16x32_bf16(
                    false, afr[fm], false, bfr[fn],
                    (short)0, acc[fm][fn], false, false);

        if (PRE) wait_asynccnt0();
        __syncthreads();
        buf ^= 1;
    }

    // ---- epilogue: C/D layout — VGPR r: lanes0-15 M=r, lanes16-31 M=8+r ----
    #pragma unroll
    for (int fn = 0; fn < 4; ++fn) {
        const int col = n0 + wn * 64 + fn * 16 + l15;
        const float bv = bias[(size_t)col * NUM_E + e];
        #pragma unroll
        for (int fm = 0; fm < 2; ++fm) {
            const int rbase = wm * 32 + fm * 16 + half * 8;
            #pragma unroll
            for (int r = 0; r < 8; ++r) {
                const int rr = rbase + r;
                if (rr < rows)
                    out[(size_t)(row0 + rr) * D_OUT + col] = acc[fm][fn][r] + bv;
            }
        }
    }
}

// =====================================================================
// Launch. ws layout: [0,17KB) tile table; [32KB, 32KB+32MB) bf16 weights.
// =====================================================================
extern "C" void kernel_launch(void* const* d_in, const int* in_sizes, int n_in,
                              void* d_out, int out_size, void* d_ws, size_t ws_size,
                              hipStream_t stream) {
    const float* x      = (const float*)d_in[0];
    const float* weight = (const float*)d_in[1];
    const float* bias   = (const float*)d_in[2];
    const int*   counts = (const int*)d_in[3];
    float*       out    = (float*)d_out;

    int4* tiles = (int4*)d_ws;
    const size_t wbfOff  = 32768;
    const size_t wbfSize = (size_t)2 * D_OUT * NUM_E * D_IN;  // 32 MB bf16

    moe_build_tiles<<<1, 32, 0, stream>>>(counts, tiles);

    dim3 grid(MAX_MTILES, NTILES);
    if (ws_size >= wbfOff + wbfSize) {
        __bf16* wbf = (__bf16*)((char*)d_ws + wbfOff);
        moe_convert_w<<<(D_OUT * NUM_E * D_IN) / (256 * 8), 256, 0, stream>>>(weight, wbf);
        moe_gemm<true><<<grid, 256, 0, stream>>>(x, weight, wbf, bias, tiles, out);
    } else {
        moe_gemm<false><<<grid, 256, 0, stream>>>(x, weight, nullptr, bias, tiles, out);
    }
}
